// MultiHeadEGATLayer_88802743812451
// MI455X (gfx1250) — matrix operand order repaired and checked by reading, not compiled, not run on hardware
//
#include <hip/hip_runtime.h>

// ---------------------------------------------------------------------------
// MultiHeadEGATLayer for MI455X (gfx1250): WMMA bf16->f32 GEMMs + TDM weight
// staging (tensor_load_to_lds) + atomic segment-softmax / scatter.
// ---------------------------------------------------------------------------
#define NN      50000     // nodes
#define EE      800000    // edges
#define DIN     64        // node/edge input feature dim
#define KE      192       // 2*DIN + DIN (edge GEMM K)
#define COLS    128       // H * DOUT = 4*32
#define NEG_SL  0.01f

typedef __attribute__((ext_vector_type(16))) __bf16 bf16x16;
typedef __attribute__((ext_vector_type(8)))  __bf16 bf16x8;
typedef __attribute__((ext_vector_type(8)))  float  floatx8;
typedef __attribute__((ext_vector_type(4)))  unsigned u32x4;
typedef __attribute__((ext_vector_type(8)))  unsigned u32x8;

union AU { bf16x16 v; bf16x8 h2[2]; unsigned short u[16]; };

__device__ __forceinline__ unsigned short f2bf(float x) {
    unsigned u = __float_as_uint(x);
    unsigned r = u + 0x7FFFu + ((u >> 16) & 1u);   // RNE
    return (unsigned short)(r >> 16);
}

__device__ __forceinline__ void cvt8(const float* __restrict__ p, unsigned short* o) {
    float4 q0 = ((const float4*)p)[0];
    float4 q1 = ((const float4*)p)[1];
    o[0]=f2bf(q0.x); o[1]=f2bf(q0.y); o[2]=f2bf(q0.z); o[3]=f2bf(q0.w);
    o[4]=f2bf(q1.x); o[5]=f2bf(q1.y); o[6]=f2bf(q1.z); o[7]=f2bf(q1.w);
}

// order-preserving float -> uint mapping (for atomicMax on signed floats)
__device__ __forceinline__ unsigned fkey(float x) {
    unsigned b = __float_as_uint(x);
    return (b & 0x80000000u) ? ~b : (b | 0x80000000u);
}
__device__ __forceinline__ float funkey(unsigned u) {
    return (u & 0x80000000u) ? __uint_as_float(u ^ 0x80000000u)
                             : __uint_as_float(~u);
}

// ---------------------------------------------------------------------------
// TDM: DMA `bytes` (multiple of 8, tile <= 64K units) global -> LDS.
// D# per CDNA5 ISA ch.7/8: 1-D tile, data_size=3 (8B units), count=1, type=2.
// Issue from ONE wave; caller waits TENSORcnt==0 then workgroup-barriers.
// ---------------------------------------------------------------------------
__device__ __forceinline__ void tdm_load_lds(unsigned lds_off, const void* gptr,
                                             unsigned bytes) {
    unsigned long long ga = (unsigned long long)gptr;
    unsigned n8 = bytes >> 3;                         // 8-byte elements
    u32x4 g0;
    g0[0] = 1u;                                       // count=1 (valid user D#)
    g0[1] = lds_off;                                  // lds_addr[31:0]
    g0[2] = (unsigned)(ga & 0xFFFFFFFFu);             // global_addr[31:0]
    g0[3] = ((unsigned)(ga >> 32) & 0x01FFFFFFu)      // global_addr[56:32]
            | 0x80000000u;                            // type=2 (bits 127:126)
    u32x8 g1;
    g1[0] = 0x00030000u;                              // wg_mask=0, data_size=3
    g1[1] = (n8 & 0xFFFFu) << 16;                     // tensor_dim0[15:0]
    g1[2] = (n8 >> 16) | (1u << 16);                  // tensor_dim0[31:16], tensor_dim1=1
    g1[3] = (n8 & 0xFFFFu) << 16;                     // tile_dim0 = n8
    g1[4] = 1u;                                       // tile_dim1=1, tile_dim2=0
    g1[5] = n8;                                       // tensor_dim0_stride[31:0]
    g1[6] = 0u;                                       // stride hi, dim1_stride lo
    g1[7] = 0u;
    u32x4 gz = {0u, 0u, 0u, 0u};                      // groups 2/3: zero (dims unused)
    asm volatile("tensor_load_to_lds %0, %1, %2, %3"
                 :: "s"(g0), "s"(g1), "s"(gz), "s"(gz)
                 : "memory");
}

// ---------------------------------------------------------------------------
// One-time weight prep: bf16 col-major panels + w_sum[f] = sum_k W_attn[f,k]
// ---------------------------------------------------------------------------
__global__ void wprep_kernel(const float* __restrict__ W_edges,
                             const float* __restrict__ W_nodes,
                             const float* __restrict__ W_attn,
                             unsigned short* __restrict__ wE,
                             unsigned short* __restrict__ wN,
                             float* __restrict__ w_sum) {
    int i = blockIdx.x * blockDim.x + threadIdx.x;
    if (i < KE * COLS) {                               // W_edges[k*128+n] -> wE[n*192+k]
        int k = i >> 7, n = i & 127;
        wE[n * KE + k] = f2bf(W_edges[i]);
    }
    int j = i - KE * COLS;
    if (j >= 0 && j < DIN * COLS) {                    // W_nodes[k*128+n] -> wN[n*64+k]
        int k = j >> 7, n = j & 127;
        wN[n * DIN + k] = f2bf(W_nodes[j]);
    }
    if (i < 32) {
        float s = 0.f;
        #pragma unroll
        for (int k = 0; k < 4; ++k) s += W_attn[i * 4 + k];
        w_sum[i] = s;
    }
}

// ---------------------------------------------------------------------------
// zero h_out, node_max keys, denom
// ---------------------------------------------------------------------------
__global__ void init_kernel(float* __restrict__ h_out, unsigned* __restrict__ node_max,
                            float* __restrict__ denom) {
    int i = blockIdx.x * blockDim.x + threadIdx.x;
    if (i < NN * COLS) h_out[i] = 0.f;
    if (i < NN * 4) { node_max[i] = 0u; denom[i] = 0.f; }
}

// ---------------------------------------------------------------------------
// h_t = nfeats @ W_nodes + b_nodes      [N,64] x [64,128] -> [N,128]
// 256 threads = 8 waves; block owns 16 rows, wave w owns column tile w.
// Weight panel DMA'd to LDS by TDM, overlapped with A staging.
// ---------------------------------------------------------------------------
__global__ __launch_bounds__(256) void node_gemm_kernel(
    const float* __restrict__ nfeats, const unsigned short* __restrict__ wN,
    const float* __restrict__ b_nodes, float* __restrict__ h_t)
{
    __shared__ alignas(128) unsigned short sW[COLS * DIN];  // 16 KB bf16 col-major [n][k]
    __shared__ alignas(64)  unsigned short sA[16 * DIN];    // 2 KB bf16 row-major [m][k]
    const int tid  = threadIdx.x;
    const int wave = tid >> 5, lane = tid & 31;
    const int ln = lane & 15, hf = lane >> 4;
    const long long rbase = (long long)blockIdx.x * 16;

    if (wave == 0)                                          // async DMA weights -> LDS
        tdm_load_lds((unsigned)(unsigned long long)(const void*)sW, wN, COLS * DIN * 2);

    for (int i = tid; i < 16 * DIN; i += 256) {             // stage A rows (bf16)
        int r = i >> 6, k = i & 63;
        sA[i] = f2bf(nfeats[(rbase + r) * DIN + k]);
    }
    if (wave == 0) __builtin_amdgcn_s_wait_tensorcnt(0);
    __syncthreads();

    // A per lane: row m=ln; elems 0..7: k=kb+8*hf.., elems 8..15: k=kb+16+8*hf..
    AU At[2];
    #pragma unroll
    for (int kt = 0; kt < 2; ++kt) {
        int kb = kt * 32;
        At[kt].h2[0] = *(const bf16x8*)&sA[ln * DIN + kb + 8 * hf];
        At[kt].h2[1] = *(const bf16x8*)&sA[ln * DIN + kb + 16 + 8 * hf];
    }

    const int col = wave * 16 + ln;
    floatx8 c = {};
    #pragma unroll
    for (int kt = 0; kt < 2; ++kt) {
        AU Bu;
        Bu.h2[0] = *(const bf16x8*)&sW[col * DIN + kt * 32 + 8 * hf];
        Bu.h2[1] = *(const bf16x8*)&sW[col * DIN + kt * 32 + 16 + 8 * hf];
        c = __builtin_amdgcn_wmma_f32_16x16x32_bf16(false, At[kt].v, false, Bu.v,
                                                    (short)0, c, false, false);
    }
    float bias = b_nodes[col];
    #pragma unroll
    for (int r = 0; r < 8; ++r)
        h_t[(rbase + r + 8 * hf) * COLS + col] = c[r] + bias;  // C/D row r (+8 hi half)
}

// ---------------------------------------------------------------------------
// f_out = leakyrelu([h_src|e|h_dst] @ W_edges + b_edges), a[e,h] = f_out . w_sum
// 256 threads = 8 waves; each wave owns 16 edges (A panel in registers in WMMA
// layout). 48 KB bf16 weight panel DMA'd once per block by TDM. 48 WMMAs/wave.
// ---------------------------------------------------------------------------
__global__ __launch_bounds__(256) void edge_gemm_kernel(
    const float* __restrict__ nfeats, const float* __restrict__ efeats,
    const int* __restrict__ src, const int* __restrict__ dst,
    const unsigned short* __restrict__ wE, const float* __restrict__ b_edges,
    const float* __restrict__ w_sum,
    float* __restrict__ f_out, float* __restrict__ a_out)
{
    __shared__ alignas(128) unsigned short sW[COLS * KE];   // 48 KB bf16 col-major [n][k]
    const int tid  = threadIdx.x;
    const int wave = tid >> 5, lane = tid & 31;
    const int ln = lane & 15, hf = lane >> 4;
    const long long ebase = ((long long)blockIdx.x * 8 + wave) * 16;

    if (wave == 0)                                          // async DMA weights -> LDS
        tdm_load_lds((unsigned)(unsigned long long)(const void*)sW, wE, COLS * KE * 2);

    // Gather this lane's stack row (edge ebase+ln) into WMMA A layout (registers).
    const long long e = ebase + ln;
    const int s = src[e], d = dst[e];
    const float* ps = nfeats + (long long)s * DIN;
    const float* pe = efeats + e * DIN;
    const float* pd = nfeats + (long long)d * DIN;
    AU At[6];
    #pragma unroll
    for (int kt = 0; kt < 6; ++kt) {
        int k0 = kt * 32 + 8 * hf;          // 8 consecutive k, inside one region
        int k1 = k0 + 16;
        const float* p0 = (k0 < 64) ? ps + k0 : (k0 < 128) ? pe + (k0 - 64) : pd + (k0 - 128);
        const float* p1 = (k1 < 64) ? ps + k1 : (k1 < 128) ? pe + (k1 - 64) : pd + (k1 - 128);
        cvt8(p0, At[kt].u);
        cvt8(p1, At[kt].u + 8);
    }
    if (wave == 0) __builtin_amdgcn_s_wait_tensorcnt(0);
    __syncthreads();

    float a_acc[8];
    #pragma unroll
    for (int nt = 0; nt < 8; ++nt) {
        const int col = nt * 16 + ln;
        floatx8 c = {};
        #pragma unroll
        for (int kt = 0; kt < 6; ++kt) {
            AU Bu;
            Bu.h2[0] = *(const bf16x8*)&sW[col * KE + kt * 32 + 8 * hf];
            Bu.h2[1] = *(const bf16x8*)&sW[col * KE + kt * 32 + 16 + 8 * hf];
            c = __builtin_amdgcn_wmma_f32_16x16x32_bf16(false, At[kt].v, false, Bu.v,
                                                        (short)0, c, false, false);
        }
        const float bias = b_edges[col];
        const float ws   = w_sum[col & 31];
        float part[8];
        #pragma unroll
        for (int r = 0; r < 8; ++r) {
            float v = c[r] + bias;
            v = (v >= 0.f) ? v : NEG_SL * v;               // leaky relu
            long long er = ebase + r + 8 * hf;             // C/D row mapping
            f_out[er * COLS + col] = v;
            part[r] = v * ws;
        }
        // reduce over the 16 columns of this tile (within each 16-lane half)
        #pragma unroll
        for (int r = 0; r < 8; ++r) {
            part[r] += __shfl_xor(part[r], 1, 32);
            part[r] += __shfl_xor(part[r], 2, 32);
            part[r] += __shfl_xor(part[r], 4, 32);
            part[r] += __shfl_xor(part[r], 8, 32);
        }
        if ((nt & 1) == 0) {
            #pragma unroll
            for (int r = 0; r < 8; ++r) a_acc[r] = part[r];
        } else {
            #pragma unroll
            for (int r = 0; r < 8; ++r) a_acc[r] += part[r];
            if (ln == 0) {                                  // lanes 0 & 16 hold full sums
                int h = nt >> 1;
                #pragma unroll
                for (int r = 0; r < 8; ++r)
                    a_out[(ebase + r + 8 * hf) * 4 + h] = a_acc[r];
            }
        }
    }
}

// ---------------------------------------------------------------------------
// segment max over dst (order-preserving uint atomicMax)
// ---------------------------------------------------------------------------
__global__ void amax_kernel(const float* __restrict__ a, const int* __restrict__ dst,
                            unsigned* __restrict__ node_max) {
    int i = blockIdx.x * blockDim.x + threadIdx.x;
    if (i >= EE * 4) return;
    int e = i >> 2, h = i & 3;
    atomicMax(&node_max[dst[e] * 4 + h], fkey(a[i]));
}

// ---------------------------------------------------------------------------
// a_exp = exp(a - max[dst]); denom[dst] += a_exp
// ---------------------------------------------------------------------------
__global__ void expsum_kernel(const float* __restrict__ a, const int* __restrict__ dst,
                              const unsigned* __restrict__ node_max,
                              float* __restrict__ a_exp, float* __restrict__ denom) {
    int i = blockIdx.x * blockDim.x + threadIdx.x;
    if (i >= EE * 4) return;
    int e = i >> 2, h = i & 3;
    float m = funkey(node_max[dst[e] * 4 + h]);
    float v = expf(a[i] - m);
    a_exp[i] = v;
    atomicAdd(&denom[dst[e] * 4 + h], v);
}

// ---------------------------------------------------------------------------
// h_out[dst] += (a_exp/denom[dst]) * h_t[src]
// ---------------------------------------------------------------------------
__global__ void scatter_kernel(const float* __restrict__ a_exp, const float* __restrict__ denom,
                               const float* __restrict__ h_t,
                               const int* __restrict__ src, const int* __restrict__ dst,
                               float* __restrict__ h_out) {
    long long i = (long long)blockIdx.x * blockDim.x + threadIdx.x;
    if (i >= (long long)EE * COLS) return;
    int e = (int)(i >> 7), c = (int)(i & 127), h = c >> 5;
    int d = dst[e], s = src[e];
    float alpha = a_exp[e * 4 + h] / denom[d * 4 + h];
    atomicAdd(&h_out[(long long)d * COLS + c], alpha * h_t[(long long)s * COLS + c]);
}

// ---------------------------------------------------------------------------
extern "C" void kernel_launch(void* const* d_in, const int* in_sizes, int n_in,
                              void* d_out, int out_size, void* d_ws, size_t ws_size,
                              hipStream_t stream) {
    const float* nfeats  = (const float*)d_in[0];
    const float* efeats  = (const float*)d_in[1];
    const int*   src     = (const int*)  d_in[2];
    const int*   dst     = (const int*)  d_in[3];
    const float* W_nodes = (const float*)d_in[4];
    const float* b_nodes = (const float*)d_in[5];
    const float* W_edges = (const float*)d_in[6];
    const float* b_edges = (const float*)d_in[7];
    const float* W_attn  = (const float*)d_in[8];

    float* h_out = (float*)d_out;                       // [N, H, D]  = N*128
    float* f_out = h_out + (long long)NN * COLS;        // [E, H, F]  = E*128

    // workspace layout
    float*    h_t      = (float*)d_ws;                          // N*128 f32
    float*    a        = h_t  + (long long)NN * COLS;           // E*4   f32
    float*    a_exp    = a    + (long long)EE * 4;              // E*4   f32
    unsigned* node_max = (unsigned*)(a_exp + (long long)EE * 4);// N*4   u32
    float*    denom    = (float*)(node_max + (long long)NN * 4);// N*4   f32
    float*    w_sum    = denom + (long long)NN * 4;             // 32    f32
    unsigned short* wE = (unsigned short*)(w_sum + 32);         // 128*192 bf16
    unsigned short* wN = wE + KE * COLS;                        // 128*64  bf16

    wprep_kernel<<<(KE * COLS + DIN * COLS + 255) / 256, 256, 0, stream>>>(
        W_edges, W_nodes, W_attn, wE, wN, w_sum);
    init_kernel<<<(NN * COLS + 255) / 256, 256, 0, stream>>>(h_out, node_max, denom);
    node_gemm_kernel<<<NN / 16, 256, 0, stream>>>(nfeats, wN, b_nodes, h_t);
    edge_gemm_kernel<<<EE / 128, 256, 0, stream>>>(nfeats, efeats, src, dst,
                                                   wE, b_edges, w_sum, f_out, a);
    amax_kernel<<<(EE * 4 + 255) / 256, 256, 0, stream>>>(a, dst, node_max);
    expsum_kernel<<<(EE * 4 + 255) / 256, 256, 0, stream>>>(a, dst, node_max, a_exp, denom);
    scatter_kernel<<<(int)(((long long)EE * COLS + 255) / 256), 256, 0, stream>>>(
        a_exp, denom, h_t, src, dst, h_out);
}
